// DeformableTransformer_17282948399173
// MI455X (gfx1250) — compile-verified, matrix operand order checked by reading, not run on hardware
//
#include <hip/hip_runtime.h>

// ---------------- types ----------------
typedef __bf16 bf16_t;
typedef __attribute__((ext_vector_type(16))) __bf16 bf16x16;
typedef __attribute__((ext_vector_type(8)))  __bf16 bf16x8;
typedef __attribute__((ext_vector_type(8)))  float  f32x8;
typedef __attribute__((ext_vector_type(4))) unsigned int u32x4;
typedef __attribute__((ext_vector_type(8))) int          i32x8;
typedef __attribute__((ext_vector_type(4))) int          i32x4;

// ---------------- problem constants ----------------
#define LEN_T   13294
#define B_T     2
#define D_T     256
#define NH_T    8
#define DH_T    32
#define DFF_T   1024
#define M_T     (B_T * LEN_T)       // 26588 rows
#define KCHUNK  256                 // K staging chunk (32KB LDS for 64 cols)

// ================= elementwise kernels =================
__global__ __launch_bounds__(256) void k_copy(float* __restrict__ dst,
                                              const float* __restrict__ src, int n) {
  int i = blockIdx.x * blockDim.x + threadIdx.x;
  if (i < n) dst[i] = src[i];
}

__global__ __launch_bounds__(256) void k_addpos_cvt(const float* __restrict__ x,
                                                    const float* __restrict__ pos,
                                                    bf16_t* __restrict__ qbf,
                                                    bf16_t* __restrict__ xbf, int n) {
  int i = blockIdx.x * blockDim.x + threadIdx.x;
  if (i < n) {
    float xv = x[i];
    qbf[i] = (bf16_t)(xv + pos[i]);
    xbf[i] = (bf16_t)xv;
  }
}

__global__ __launch_bounds__(256) void k_cvt(const float* __restrict__ in,
                                             bf16_t* __restrict__ out, int n) {
  int i = blockIdx.x * blockDim.x + threadIdx.x;
  if (i < n) out[i] = (bf16_t)in[i];
}

// weights: in [K,N] fp32 row-major -> out [N,K] bf16 row-major (transposed)
__global__ __launch_bounds__(256) void k_wt_cvt(const float* __restrict__ in,
                                                bf16_t* __restrict__ out, int K, int N) {
  int i = blockIdx.x * blockDim.x + threadIdx.x;
  if (i < K * N) {
    int k = i / N, nn = i % N;
    out[(size_t)nn * K + k] = (bf16_t)in[i];
  }
}

// ================= WMMA bf16 GEMM with TDM-staged weight tiles =================
// Block (8 waves) computes a 128x64 C macro-tile.
// B (weights, [N,K] bf16) chunk 64x256 is DMA'd into LDS by the Tensor Data Mover,
// then all 8 waves reuse it from LDS; A fragments stream from global.
// Each wave: 1 A-frag x 4 B-frags -> 4 16x16 f32 accumulators per K-step of 32.
__global__ __launch_bounds__(256) void k_gemm(const bf16_t* __restrict__ A,
                                              const bf16_t* __restrict__ Bt,
                                              const float* __restrict__ bias,
                                              float* __restrict__ Cf,
                                              bf16_t* __restrict__ Cbf,
                                              int Mdim, int Ndim, int Kdim, int doRelu) {
  __shared__ bf16_t ldsB[64 * KCHUNK];          // 32 KB

  const int tilesN64 = Ndim >> 6;
  const int bm = blockIdx.x / tilesN64;
  const int bn = blockIdx.x % tilesN64;
  const int waveId = threadIdx.x >> 5;
  const int lane = threadIdx.x & 31;
  const int half = lane >> 4;                   // 0: K 0..7 & 16..23, 1: K 8..15 & 24..31
  const int lidx = lane & 15;
  const int col0 = bn * 64;

  int arow = bm * 128 + waveId * 16 + lidx;
  if (arow >= Mdim) arow = Mdim - 1;            // clamp (edge rows not stored)
  const bf16_t* aptr = A + (size_t)arow * Kdim + half * 8;

  const unsigned int ldsBase = (unsigned int)(uintptr_t)(&ldsB[0]);  // LDS byte offset

  f32x8 acc[4] = {};

  for (int kc = 0; kc < Kdim; kc += KCHUNK) {
    // ---- stage Bt[col0 .. col0+63][kc .. kc+KCHUNK-1] into LDS via TDM ----
    if (threadIdx.x < 32) {
      const unsigned long long gaddr =
          (unsigned long long)(uintptr_t)(Bt + (size_t)col0 * Kdim + kc);
      u32x4 g0;
      g0[0] = 1u;                                            // count=1, user mode
      g0[1] = ldsBase;                                       // lds_addr (bytes)
      g0[2] = (unsigned int)gaddr;                           // global_addr[31:0]
      g0[3] = (unsigned int)(gaddr >> 32) | (2u << 30);      // global_addr[56:32] | type=2
      i32x8 g1;
      g1[0] = 0x10000;                                       // data_size=1 (2 bytes)
      g1[1] = (int)(((unsigned)Kdim & 0xFFFFu) << 16);       // tensor_dim0[15:0]
      g1[2] = (int)((((unsigned)Kdim >> 16) & 0xFFFFu) | (64u << 16)); // dim0 hi | tensor_dim1=64
      g1[3] = (int)((unsigned)KCHUNK << 16);                 // tile_dim0 = KCHUNK
      g1[4] = 64;                                            // tile_dim1 = 64 (tile_dim2=0)
      g1[5] = Kdim;                                          // tensor_dim0_stride[31:0]
      g1[6] = 0;                                             // stride hi | dim1_stride lo
      g1[7] = 0;
      i32x4 g2 = {0, 0, 0, 0};
      i32x4 g3 = {0, 0, 0, 0};
#if __clang_major__ >= 23
      i32x8 g4 = {};
      __builtin_amdgcn_tensor_load_to_lds(g0, g1, g2, g3, g4, 0);
#else
      __builtin_amdgcn_tensor_load_to_lds(g0, g1, g2, g3, 0);
#endif
      __builtin_amdgcn_s_wait_tensorcnt(0);
    }
    __syncthreads();

    // ---- compute on the staged chunk ----
    const bf16_t* ap = aptr + kc;
    for (int k0 = 0; k0 < KCHUNK; k0 += 32) {
      bf16x8 alo = *reinterpret_cast<const bf16x8*>(ap + k0);
      bf16x8 ahi = *reinterpret_cast<const bf16x8*>(ap + k0 + 16);
      bf16x16 a = __builtin_shufflevector(alo, ahi, 0,1,2,3,4,5,6,7,8,9,10,11,12,13,14,15);
#pragma unroll
      for (int j = 0; j < 4; ++j) {
        const int cj = j * 16 + lidx;                       // column within 64-wide tile
        const bf16_t* bp = &ldsB[cj * KCHUNK + k0 + half * 8];
        bf16x8 blo = *reinterpret_cast<const bf16x8*>(bp);
        bf16x8 bhi = *reinterpret_cast<const bf16x8*>(bp + 16);
        bf16x16 b = __builtin_shufflevector(blo, bhi, 0,1,2,3,4,5,6,7,8,9,10,11,12,13,14,15);
        acc[j] = __builtin_amdgcn_wmma_f32_16x16x32_bf16(false, a, false, b,
                                                         (short)0, acc[j], false, false);
      }
    }
    __syncthreads();   // protect LDS before next chunk restage
  }

  // ---- epilogue: bias, optional ReLU, f32 or bf16 store ----
#pragma unroll
  for (int j = 0; j < 4; ++j) {
    const int col = col0 + j * 16 + lidx;
    const float bv = bias ? bias[col] : 0.0f;
#pragma unroll
    for (int r = 0; r < 8; ++r) {
      const int row = bm * 128 + waveId * 16 + half * 8 + r;
      if (row < Mdim) {
        float v = acc[j][r] + bv;
        if (doRelu) v = v > 0.0f ? v : 0.0f;
        if (Cbf) Cbf[(size_t)row * Ndim + col] = (bf16_t)v;
        else     Cf [(size_t)row * Ndim + col] = v;
      }
    }
  }
}

// ================= softmax over 16 attn logits per (b,lq,head) =================
__global__ __launch_bounds__(256) void k_softmax16(float* __restrict__ attn, int rows) {
  int r = blockIdx.x * blockDim.x + threadIdx.x;
  if (r >= rows) return;
  int t = r >> 3, nh = r & 7;
  float* p = attn + (size_t)t * 128 + nh * 16;
  float mx = p[0];
#pragma unroll
  for (int i = 1; i < 16; ++i) mx = fmaxf(mx, p[i]);
  float e[16]; float s = 0.0f;
#pragma unroll
  for (int i = 0; i < 16; ++i) { e[i] = __expf(p[i] - mx); s += e[i]; }
  float inv = 1.0f / s;
#pragma unroll
  for (int i = 0; i < 16; ++i) p[i] = e[i] * inv;
}

// ================= multi-scale deformable attention sampling =================
// one wave per (b, lq, nh); lane = channel (DH=32) -> coalesced value gathers
__global__ __launch_bounds__(256) void k_msda(const float* __restrict__ value,
                                              const float* __restrict__ off,
                                              const float* __restrict__ attn,
                                              float* __restrict__ out) {
  const int ROWS = M_T * NH_T;
  const int wave = blockIdx.x * 8 + (threadIdx.x >> 5);
  if (wave >= ROWS) return;
  const int lane = threadIdx.x & 31;
  const int nh = wave & 7;
  const int t  = wave >> 3;            // b*LEN + lq
  const int lq = t % LEN_T;
  const int b  = t / LEN_T;

  // reference point (grid-cell center of lq's own level; same for all sampling levels)
  float rx, ry;
  if (lq < 10000)      { int i = lq;         rx = ((i % 100) + 0.5f) * (1.0f/100.0f); ry = ((i / 100) + 0.5f) * (1.0f/100.0f); }
  else if (lq < 12500) { int i = lq - 10000; rx = ((i %  50) + 0.5f) * (1.0f/ 50.0f); ry = ((i /  50) + 0.5f) * (1.0f/ 50.0f); }
  else if (lq < 13125) { int i = lq - 12500; rx = ((i %  25) + 0.5f) * (1.0f/ 25.0f); ry = ((i /  25) + 0.5f) * (1.0f/ 25.0f); }
  else                 { int i = lq - 13125; rx = ((i %  13) + 0.5f) * (1.0f/ 13.0f); ry = ((i /  13) + 0.5f) * (1.0f/ 13.0f); }

  const int hs[4] = {100, 50, 25, 13};
  const int wsz[4]= {100, 50, 25, 13};
  const int st[4] = {0, 10000, 12500, 13125};

  const size_t orow = (size_t)t * 256;   // offset row (N=256)
  const size_t arow = (size_t)t * 128;   // attn row   (N=128)
  float acc = 0.0f;

#pragma unroll
  for (int l = 0; l < 4; ++l) {
    const float wlf = (float)wsz[l], hlf = (float)hs[l];
#pragma unroll
    for (int p = 0; p < 4; ++p) {
      const int oc = ((nh * 4 + l) * 4 + p) * 2;
      const float ox = off[orow + oc];
      const float oy = off[orow + oc + 1];
      const float aw = attn[arow + (nh * 4 + l) * 4 + p];
      const float sx = (rx + ox / wlf) * wlf - 0.5f;
      const float sy = (ry + oy / hlf) * hlf - 0.5f;
      const float x0f = floorf(sx), y0f = floorf(sy);
      const float wx = sx - x0f, wy = sy - y0f;
      const int x0 = (int)x0f, y0 = (int)y0f;
      const float cw[4] = { (1.0f-wx)*(1.0f-wy), wx*(1.0f-wy), (1.0f-wx)*wy, wx*wy };
      const int dxs[4] = {0,1,0,1}, dys[4] = {0,0,1,1};
#pragma unroll
      for (int cc = 0; cc < 4; ++cc) {
        const int xc = x0 + dxs[cc], yc = y0 + dys[cc];
        if (xc >= 0 && xc < wsz[l] && yc >= 0 && yc < hs[l]) {   // wave-uniform branch
          const int pix = st[l] + yc * wsz[l] + xc;
          const float g = value[(((size_t)b * LEN_T + pix) * NH_T + nh) * DH_T + lane];
          acc += g * (cw[cc] * aw);
        }
      }
    }
  }
  out[(size_t)wave * DH_T + lane] = acc;   // [B,LEN,NH,DH] == [M,256]
}

// ================= residual + LayerNorm (wave per row of 256) =================
__global__ __launch_bounds__(256) void k_ln(float* __restrict__ x,
                                            const float* __restrict__ y,
                                            const float* __restrict__ g,
                                            const float* __restrict__ bb, int rows) {
  const int row = blockIdx.x * 8 + (threadIdx.x >> 5);
  if (row >= rows) return;
  const int lane = threadIdx.x & 31;
  float v[8]; float s = 0.0f;
#pragma unroll
  for (int i = 0; i < 8; ++i) {
    const int idx = i * 32 + lane;
    v[i] = x[(size_t)row * 256 + idx] + y[(size_t)row * 256 + idx];
    s += v[i];
  }
#pragma unroll
  for (int o = 16; o > 0; o >>= 1) s += __shfl_xor(s, o, 32);
  const float mean = s * (1.0f / 256.0f);
  float vs = 0.0f;
#pragma unroll
  for (int i = 0; i < 8; ++i) { float d = v[i] - mean; vs += d * d; }
#pragma unroll
  for (int o = 16; o > 0; o >>= 1) vs += __shfl_xor(vs, o, 32);
  const float rinv = rsqrtf(vs * (1.0f / 256.0f) + 1e-5f);
#pragma unroll
  for (int i = 0; i < 8; ++i) {
    const int idx = i * 32 + lane;
    x[(size_t)row * 256 + idx] = (v[i] - mean) * rinv * g[idx] + bb[idx];
  }
}

// ================= host launch =================
extern "C" void kernel_launch(void* const* d_in, const int* in_sizes, int n_in,
                              void* d_out, int out_size, void* d_ws, size_t ws_size,
                              hipStream_t stream) {
  (void)in_sizes; (void)n_in; (void)out_size; (void)ws_size;

  const float* src = (const float*)d_in[0];
  const float* pos = (const float*)d_in[1];
  auto P = [&](int layer, int idx) -> const float* {
    return (const float*)d_in[2 + layer * 16 + idx];
  };
  // param idx: 0 Wv,1 bv,2 Woff,3 boff,4 Wat,5 bat,6 Wo,7 bo,
  //            8 ln1_g,9 ln1_b,10 W1,11 b1,12 W2,13 b2,14 ln2_g,15 ln2_b

  char* ws = (char*)d_ws;
  constexpr size_t SZ_F256  = (size_t)M_T * 256 * 4;   // 27.2 MB
  constexpr size_t SZ_F128  = (size_t)M_T * 128 * 4;   // 13.6 MB
  constexpr size_t SZ_BF256 = (size_t)M_T * 256 * 2;   // 13.6 MB

  float*  value = (float*)(ws);
  float*  offb  = (float*)(ws + SZ_F256);
  float*  attn  = (float*)(ws + 2 * SZ_F256);
  float*  msda  = (float*)(ws + 2 * SZ_F256 + SZ_F128);
  float*  ybuf  = (float*)(ws + 3 * SZ_F256 + SZ_F128);
  bf16_t* qbf   = (bf16_t*)(ws + 4 * SZ_F256 + SZ_F128);
  bf16_t* xbf   = (bf16_t*)(ws + 4 * SZ_F256 + SZ_F128 + SZ_BF256);
  char*   wbase =          ws + 4 * SZ_F256 + SZ_F128 + 2 * SZ_BF256;
  bf16_t* wv_t   = (bf16_t*)wbase;
  bf16_t* woff_t = wv_t   + 256 * 256;
  bf16_t* wat_t  = woff_t + 256 * 256;
  bf16_t* wo_t   = wat_t  + 256 * 128;
  bf16_t* w1_t   = wo_t   + 256 * 256;
  bf16_t* w2_t   = w1_t   + 256 * 1024;
  // aliases (phase-disjoint): FFN hidden bf16 over value+offb; msda-bf16 over attn
  bf16_t* hidden = (bf16_t*)ws;
  bf16_t* msdabf = (bf16_t*)(ws + 2 * SZ_F256);

  float* x = (float*)d_out;            // residual stream lives in d_out
  const int n   = M_T * 256;
  const int gEl = (n + 255) / 256;
  const int tilesM128 = (M_T + 127) / 128;
  auto gemmGrid = [&](int N) { return tilesM128 * (N / 64); };

  k_copy<<<gEl, 256, 0, stream>>>(x, src, n);

  for (int L = 0; L < 6; ++L) {
    k_addpos_cvt<<<gEl, 256, 0, stream>>>(x, pos, qbf, xbf, n);

    k_wt_cvt<<<(256 * 256 + 255) / 256, 256, 0, stream>>>(P(L, 0),  wv_t,   256, 256);
    k_wt_cvt<<<(256 * 256 + 255) / 256, 256, 0, stream>>>(P(L, 2),  woff_t, 256, 256);
    k_wt_cvt<<<(256 * 128 + 255) / 256, 256, 0, stream>>>(P(L, 4),  wat_t,  256, 128);
    k_wt_cvt<<<(256 * 256 + 255) / 256, 256, 0, stream>>>(P(L, 6),  wo_t,   256, 256);
    k_wt_cvt<<<(256 * 1024 + 255) / 256, 256, 0, stream>>>(P(L, 10), w1_t,  256, 1024);
    k_wt_cvt<<<(1024 * 256 + 255) / 256, 256, 0, stream>>>(P(L, 12), w2_t,  1024, 256);

    // value / offsets / attn-logits projections (WMMA + TDM-staged weights)
    k_gemm<<<gemmGrid(256), 256, 0, stream>>>(xbf, wv_t,   P(L, 1), value, (bf16_t*)nullptr, M_T, 256, 256, 0);
    k_gemm<<<gemmGrid(256), 256, 0, stream>>>(qbf, woff_t, P(L, 3), offb,  (bf16_t*)nullptr, M_T, 256, 256, 0);
    k_gemm<<<gemmGrid(128), 256, 0, stream>>>(qbf, wat_t,  P(L, 5), attn,  (bf16_t*)nullptr, M_T, 128, 256, 0);

    k_softmax16<<<(M_T * 8 + 255) / 256, 256, 0, stream>>>(attn, M_T * 8);
    k_msda<<<(M_T * 8 + 7) / 8, 256, 0, stream>>>(value, offb, attn, msda);

    // output projection + residual LN1
    k_cvt<<<gEl, 256, 0, stream>>>(msda, msdabf, n);
    k_gemm<<<gemmGrid(256), 256, 0, stream>>>(msdabf, wo_t, P(L, 7), ybuf, (bf16_t*)nullptr, M_T, 256, 256, 0);
    k_ln<<<(M_T + 7) / 8, 256, 0, stream>>>(x, ybuf, P(L, 8), P(L, 9), M_T);

    // FFN (WMMA, hidden kept in bf16) + residual LN2
    k_cvt<<<gEl, 256, 0, stream>>>(x, xbf, n);
    k_gemm<<<gemmGrid(1024), 256, 0, stream>>>(xbf, w1_t, P(L, 11), (float*)nullptr, hidden, M_T, 1024, 256, 1);
    k_gemm<<<gemmGrid(256), 256, 0, stream>>>(hidden, w2_t, P(L, 13), ybuf, (bf16_t*)nullptr, M_T, 256, 1024, 0);
    k_ln<<<(M_T + 7) / 8, 256, 0, stream>>>(x, ybuf, P(L, 14), P(L, 15), M_T);
  }
}
